// VisionMamba_32401233281533
// MI455X (gfx1250) — compile-verified, matrix operand order checked by reading, not run on hardware
//
#include <hip/hip_runtime.h>
#include <hip/hip_bf16.h>
#include <math.h>

// ---- model dims ----
#define B_     32
#define DM     192
#define DEPTH_ 24
#define DI     384
#define DS     16
#define DTR    12
#define DC     4
#define NP     196
#define LSEQ   197
#define TPOS   98
#define NCLS   1000
#define EPS_   1e-5f
#define XPN    48      // 44 (dt_rank + 2*d_state) padded to 48 for WMMA N-tiles

typedef __attribute__((ext_vector_type(16))) __bf16 v16bf;
typedef __attribute__((ext_vector_type(8)))  float  v8f;

union AccU  { v8f v; float f[8]; };
union FragU { v16bf v; uint4 q[2]; };

__device__ __forceinline__ unsigned short to_bf16(float f) {
  unsigned int u = __float_as_uint(f);
  u += 0x7FFFu + ((u >> 16) & 1u);          // round-to-nearest-even
  return (unsigned short)(u >> 16);
}
__device__ __forceinline__ float from_bf16(unsigned short h) {
  return __uint_as_float(((unsigned int)h) << 16);
}
__device__ __forceinline__ float silu_f(float v) {
  return v / (1.f + __expf(-v));
}

// Async global -> LDS copy of one 16B chunk per lane (CDNA5 async engine,
// tracked by ASYNCcnt, bypasses VGPRs).
__device__ __forceinline__ void async_b128(unsigned lds_off,
                                           const unsigned short* g) {
  asm volatile("global_load_async_to_lds_b128 %0, %1, off"
               :: "v"(lds_off), "v"(g) : "memory");
}
__device__ __forceinline__ void wait_async0() {
  asm volatile("s_wait_asynccnt 0x0" ::: "memory");
}

// Stage a (NTILES*16) x 32 bf16 B-tile into LDS, laid out in exact WMMA
// fragment order: chunk tid -> nt = tid>>6, reader-lane = (tid&63)>>1,
// piece = tid&1.  Reader lane then reads 32 contiguous bytes per n-tile.
template<int NTILES>
__device__ __forceinline__ void stage_tile(const unsigned short* __restrict__ W,
                                           int K, int n0, int k0,
                                           unsigned lds_base, int tid) {
  if (tid < NTILES * 64) {                       // wave-uniform (multiple of 32)
    const int nt = tid >> 6;
    const int c  = tid & 63;
    const int ln = c >> 1;                       // destination reader lane 0..31
    const int pc = c & 1;
    const int rr = ln & 15, hh = ln >> 4;
    const unsigned short* g =
        W + (size_t)(n0 + nt * 16 + rr) * K + k0 + pc * 16 + hh * 8;
    async_b128(lds_base + (unsigned)(tid << 4), g);
  }
}

// ============ WMMA bf16 GEMM:  C[M,N] = A[M,K] * W[N,K]^T ====================
// A row-major bf16, W row-major bf16 (weight layout (N,K)), C fp32 row-major.
// K % 32 == 0, M % 16 == 0.  Block: 8 waves x 16 rows, NTILES*16 cols.
// B tile is async-staged to LDS (double buffered); one barrier per K-step.
// All B fragments are pulled to registers first so one s_wait_dscnt covers
// all NTILES WMMAs per K-step.
template<int NTILES>
__global__ __launch_bounds__(256) void gemm_bf16_kernel(
    const unsigned short* __restrict__ A,
    const unsigned short* __restrict__ W,
    float* __restrict__ C,
    int M, int N, int K)
{
  __shared__ __align__(16) unsigned char smem[2][4 * 1024];
  const int tid  = threadIdx.x;
  const int lane = tid & 31;
  const int wave = tid >> 5;
  const int half = lane >> 4;        // K-half selector per CDNA5 A/B layout
  const int r    = lane & 15;        // row within A tile
  int m0 = (blockIdx.y * 8 + wave) * 16;
  if (m0 + 16 > M) m0 = M - 16;      // clamp: keeps barrier participation uniform
  const int n0 = blockIdx.x * (NTILES * 16);

  AccU acc[NTILES];
  #pragma unroll
  for (int i = 0; i < NTILES; ++i)
    #pragma unroll
    for (int j = 0; j < 8; ++j) acc[i].f[j] = 0.f;

  const unsigned short* arow = A + (size_t)(m0 + r) * K;
  const unsigned lds0 = (unsigned)(uintptr_t)&smem[0][0];
  const unsigned lds1 = (unsigned)(uintptr_t)&smem[1][0];

  stage_tile<NTILES>(W, K, n0, 0, lds0, tid);        // prologue stage

  const int steps = K / 32;
  for (int s = 0; s < steps; ++s) {
    const int k0 = s * 32;
    FragU af;                                         // A frag: direct global
    af.q[0] = *(const uint4*)(arow + k0 + half * 8);
    af.q[1] = *(const uint4*)(arow + k0 + 16 + half * 8);
    if (k0 + 64 < K) __builtin_prefetch(arow + k0 + 64, 0, 3);

    wait_async0();                                    // own stage(s) complete
    __syncthreads();                                  // everyone's stage(s) visible
    if (s + 1 < steps)                                // overlap stage(s+1) w/ compute
      stage_tile<NTILES>(W, K, n0, k0 + 32, ((s + 1) & 1) ? lds1 : lds0, tid);

    const unsigned char* buf = smem[s & 1];
    FragU bf[NTILES];
    #pragma unroll
    for (int nt = 0; nt < NTILES; ++nt) {             // batched ds_loads ...
      const uint4* lb = (const uint4*)(buf + ((nt * 32 + lane) << 5));
      bf[nt].q[0] = lb[0];
      bf[nt].q[1] = lb[1];
    }
    #pragma unroll
    for (int nt = 0; nt < NTILES; ++nt) {             // ... then back-to-back WMMAs
      acc[nt].v = __builtin_amdgcn_wmma_f32_16x16x32_bf16(
          false, af.v, false, bf[nt].v, (short)0, acc[nt].v, false, false);
    }
  }

  #pragma unroll
  for (int nt = 0; nt < NTILES; ++nt) {
    const int col = n0 + nt * 16 + r;
    #pragma unroll
    for (int v = 0; v < 8; ++v) {
      const int row = m0 + v + half * 8;              // documented C layout
      C[(size_t)row * N + col] = acc[nt].f[v];
    }
  }
}

// ======================= elementwise / prep kernels ==========================
__global__ void cvt_bf16_kernel(const float* __restrict__ src,
                                unsigned short* __restrict__ dst, size_t n) {
  size_t i = (size_t)blockIdx.x * blockDim.x + threadIdx.x;
  const size_t st = (size_t)gridDim.x * blockDim.x;
  for (; i < n; i += st) dst[i] = to_bf16(src[i]);
}

// pack xproj (24,44,384) fwd/bwd -> bf16 (24,2,48,384), rows 44..47 zero
__global__ void pack_xproj_kernel(const float* __restrict__ xf,
                                  const float* __restrict__ xb,
                                  unsigned short* __restrict__ dst) {
  const size_t total = (size_t)DEPTH_ * 2 * XPN * DI;
  size_t i = (size_t)blockIdx.x * blockDim.x + threadIdx.x;
  const size_t st = (size_t)gridDim.x * blockDim.x;
  for (; i < total; i += st) {
    const size_t k   = i % DI;
    const size_t e   = (i / DI) % XPN;
    const size_t dir = (i / ((size_t)DI * XPN)) % 2;
    const size_t l   = i / ((size_t)DI * XPN * 2);
    float v = 0.f;
    if (e < (DTR + 2 * DS)) {
      const float* s = dir ? xb : xf;
      v = s[(l * (DTR + 2 * DS) + e) * DI + k];
    }
    dst[i] = to_bf16(v);
  }
}

// im2col for 16x16 stride-16 patches: (B*196, 768) bf16, K = c*256 + py*16 + px
__global__ void im2col_kernel(const float* __restrict__ x,
                              unsigned short* __restrict__ out) {
  const size_t total = (size_t)B_ * NP * 768;
  size_t i = (size_t)blockIdx.x * blockDim.x + threadIdx.x;
  const size_t st = (size_t)gridDim.x * blockDim.x;
  for (; i < total; i += st) {
    const int k   = (int)(i % 768);
    const size_t row = i / 768;
    const int b = (int)(row / NP), p = (int)(row % NP);
    const int ph = p / 14, pw = p % 14;
    const int c = k / 256, rem = k % 256;
    const int py = rem / 16, px = rem % 16;
    const size_t src = (((size_t)b * 3 + c) * 224 + (ph * 16 + py)) * 224 + (pw * 16 + px);
    out[i] = to_bf16(x[src]);
  }
}

// feat (+bias) + cls insertion + pos_embed -> hidden; residual = 0
__global__ void assemble_kernel(const float* __restrict__ featC,
                                const float* __restrict__ patch_b,
                                const float* __restrict__ cls_tok,
                                const float* __restrict__ pos_emb,
                                float* __restrict__ hidden,
                                float* __restrict__ residual) {
  const size_t total = (size_t)B_ * LSEQ * DM;
  size_t i = (size_t)blockIdx.x * blockDim.x + threadIdx.x;
  const size_t st = (size_t)gridDim.x * blockDim.x;
  for (; i < total; i += st) {
    const int c = (int)(i % DM);
    const size_t bt = i / DM;
    const int t = (int)(bt % LSEQ), b = (int)(bt / LSEQ);
    float v;
    if (t == TPOS) v = cls_tok[c];
    else {
      const int p = (t < TPOS) ? t : (t - 1);
      v = featC[((size_t)b * NP + p) * DM + c] + patch_b[c];
    }
    v += pos_emb[(size_t)t * DM + c];
    hidden[i]   = v;
    residual[i] = 0.f;
  }
}

// residual += hidden; hs_bf16 = RMSNorm(residual) * norm_w[l]
__global__ __launch_bounds__(192) void prelayer_kernel(
    float* __restrict__ residual, const float* __restrict__ hidden,
    const float* __restrict__ norm_w_l, unsigned short* __restrict__ hs_bf) {
  __shared__ float red[8];
  const size_t bt = blockIdx.x;
  const int c = threadIdx.x;
  const size_t idx = bt * DM + c;
  const float r = residual[idx] + hidden[idx];
  residual[idx] = r;
  float s = r * r;
  #pragma unroll
  for (int off = 16; off; off >>= 1) s += __shfl_xor(s, off, 32);
  if ((c & 31) == 0) red[c >> 5] = s;
  __syncthreads();
  if (c == 0) {
    float t = 0.f;
    for (int i = 0; i < 6; ++i) t += red[i];
    red[0] = rsqrtf(t / (float)DM + EPS_);
  }
  __syncthreads();
  hs_bf[idx] = to_bf16(r * red[0] * norm_w_l[c]);
}

// causal depthwise conv (D_CONV=4) + SiLU over (possibly reversed) sequence
__global__ void conv_kernel(const float* __restrict__ xz,
                            const float* __restrict__ conv_w_l,
                            const float* __restrict__ conv_b_l,
                            unsigned short* __restrict__ xconv_bf, int rev) {
  const size_t total = (size_t)B_ * LSEQ * DI;
  size_t i = (size_t)blockIdx.x * blockDim.x + threadIdx.x;
  const size_t st = (size_t)gridDim.x * blockDim.x;
  for (; i < total; i += st) {
    const int d = (int)(i % DI);
    const size_t bt = i / DI;
    const int t = (int)(bt % LSEQ), b = (int)(bt / LSEQ);
    float acc = conv_b_l[d];
    #pragma unroll
    for (int k = 0; k < DC; ++k) {
      const int tt = t - (DC - 1) + k;        // position in scan order
      if (tt >= 0) {
        const int ts = rev ? (LSEQ - 1 - tt) : tt;   // map to original order
        acc += conv_w_l[d * DC + k] * xz[((size_t)b * LSEQ + ts) * (2 * DI) + d];
      }
    }
    xconv_bf[i] = to_bf16(silu_f(acc));
  }
}

// selective scan: one lane per (b,d); dt-projection (K=12) folded in
__global__ __launch_bounds__(256) void scan_kernel(
    const unsigned short* __restrict__ xconv_bf,
    const float* __restrict__ dbc,      // (B*L, 48): [0:12)=dt-in, [12:28)=B, [28:44)=C
    const float* __restrict__ dtw_l,    // (DI, 12)
    const float* __restrict__ dtb_l,    // (DI)
    const float* __restrict__ Alog_l,   // (DI, 16)
    const float* __restrict__ D_l,      // (DI)
    float* __restrict__ yraw) {
  const int gid = blockIdx.x * 256 + threadIdx.x;
  if (gid >= B_ * DI) return;
  const int b = gid / DI, d = gid % DI;
  float dtw[DTR];
  #pragma unroll
  for (int r = 0; r < DTR; ++r) dtw[r] = dtw_l[d * DTR + r];
  const float dtb = dtb_l[d];
  float A[DS];
  #pragma unroll
  for (int j = 0; j < DS; ++j) A[j] = -__expf(Alog_l[d * DS + j]);
  const float Dp = D_l[d];
  float h[DS];
  #pragma unroll
  for (int j = 0; j < DS; ++j) h[j] = 0.f;

  for (int t = 0; t < LSEQ; ++t) {
    const float* row = dbc + ((size_t)b * LSEQ + t) * XPN;
    float dtr = dtb;
    #pragma unroll
    for (int r = 0; r < DTR; ++r) dtr += dtw[r] * row[r];
    const float dt = (dtr > 20.f) ? dtr : __logf(1.f + __expf(dtr));  // softplus
    const float xv = from_bf16(xconv_bf[((size_t)b * LSEQ + t) * DI + d]);
    const float dx = dt * xv;
    float y = 0.f;
    #pragma unroll
    for (int j = 0; j < DS; ++j) {
      h[j] = __expf(dt * A[j]) * h[j] + dx * row[DTR + j];
      y += h[j] * row[DTR + DS + j];
    }
    yraw[((size_t)b * LSEQ + t) * DI + d] = y + xv * Dp;
  }
}

// y_bf16 = 0.5*(yf[t] + yb[L-1-t]) * silu(z)
__global__ void combine_kernel(const float* __restrict__ yf,
                               const float* __restrict__ yb,
                               const float* __restrict__ xz,
                               unsigned short* __restrict__ ybf) {
  const size_t total = (size_t)B_ * LSEQ * DI;
  size_t i = (size_t)blockIdx.x * blockDim.x + threadIdx.x;
  const size_t st = (size_t)gridDim.x * blockDim.x;
  for (; i < total; i += st) {
    const int d = (int)(i % DI);
    const size_t bt = i / DI;
    const int t = (int)(bt % LSEQ), b = (int)(bt / LSEQ);
    const float z = xz[((size_t)b * LSEQ + t) * (2 * DI) + DI + d];
    const float v = 0.5f * (yf[i] + yb[((size_t)b * LSEQ + (LSEQ - 1 - t)) * DI + d]);
    ybf[i] = to_bf16(v * silu_f(z));
  }
}

__global__ __launch_bounds__(192) void final_norm_kernel(
    const float* __restrict__ hidden, const float* __restrict__ residual,
    const float* __restrict__ normf_w, float* __restrict__ ynorm) {
  __shared__ float red[8];
  const int b = blockIdx.x, c = threadIdx.x;
  const size_t idx = ((size_t)b * LSEQ + TPOS) * DM + c;
  const float v = hidden[idx] + residual[idx];
  float s = v * v;
  #pragma unroll
  for (int off = 16; off; off >>= 1) s += __shfl_xor(s, off, 32);
  if ((c & 31) == 0) red[c >> 5] = s;
  __syncthreads();
  if (c == 0) {
    float t = 0.f;
    for (int i = 0; i < 6; ++i) t += red[i];
    red[0] = rsqrtf(t / (float)DM + EPS_);
  }
  __syncthreads();
  ynorm[(size_t)b * DM + c] = v * red[0] * normf_w[c];
}

__global__ void head_kernel(const float* __restrict__ ynorm,
                            const float* __restrict__ head_w,
                            const float* __restrict__ head_b,
                            float* __restrict__ out) {
  const int gid = blockIdx.x * 256 + threadIdx.x;
  if (gid >= B_ * NCLS) return;
  const int b = gid / NCLS, n = gid % NCLS;
  const float* y = ynorm + (size_t)b * DM;
  const float* w = head_w + (size_t)n * DM;
  float acc = head_b[n];
  for (int c = 0; c < DM; ++c) acc += y[c] * w[c];
  out[gid] = acc;
}

// ================================ launcher ===================================
extern "C" void kernel_launch(void* const* d_in, const int* in_sizes, int n_in,
                              void* d_out, int out_size, void* d_ws, size_t ws_size,
                              hipStream_t stream) {
  (void)in_sizes; (void)n_in; (void)out_size; (void)ws_size;
  const float* x         = (const float*)d_in[0];
  const float* patch_w   = (const float*)d_in[1];
  const float* patch_b   = (const float*)d_in[2];
  const float* cls_tok   = (const float*)d_in[3];
  const float* pos_emb   = (const float*)d_in[4];
  const float* norm_w    = (const float*)d_in[5];
  const float* in_proj   = (const float*)d_in[6];
  const float* conv_wf   = (const float*)d_in[7];
  const float* conv_bf_  = (const float*)d_in[8];
  const float* xproj_f   = (const float*)d_in[9];
  const float* dtw_f     = (const float*)d_in[10];
  const float* dtb_f     = (const float*)d_in[11];
  const float* Alog_f    = (const float*)d_in[12];
  const float* D_f       = (const float*)d_in[13];
  const float* conv_wb   = (const float*)d_in[14];
  const float* conv_bb_  = (const float*)d_in[15];
  const float* xproj_b   = (const float*)d_in[16];
  const float* dtw_b     = (const float*)d_in[17];
  const float* dtb_b     = (const float*)d_in[18];
  const float* Alog_b    = (const float*)d_in[19];
  const float* D_b       = (const float*)d_in[20];
  const float* out_proj  = (const float*)d_in[21];
  const float* normf_w   = (const float*)d_in[22];
  const float* head_w    = (const float*)d_in[23];
  const float* head_b    = (const float*)d_in[24];

  char* ws = (char*)d_ws;
  size_t off = 0;
  auto alloc = [&](size_t bytes) -> char* {
    char* p = ws + off;
    off += (bytes + 255) & ~(size_t)255;
    return p;
  };

  const int MT = B_ * LSEQ;  // 6304 tokens
  unsigned short* wbf_inproj  = (unsigned short*)alloc((size_t)DEPTH_ * 2 * DI * DM * 2);
  unsigned short* wbf_outproj = (unsigned short*)alloc((size_t)DEPTH_ * DM * DI * 2);
  unsigned short* wbf_patch   = (unsigned short*)alloc((size_t)DM * 768 * 2);
  unsigned short* wbf_xproj   = (unsigned short*)alloc((size_t)DEPTH_ * 2 * XPN * DI * 2);
  unsigned short* im2c        = (unsigned short*)alloc((size_t)B_ * NP * 768 * 2);
  float* featC   = (float*)alloc((size_t)B_ * NP * DM * 4);
  float* hidden  = (float*)alloc((size_t)MT * DM * 4);
  float* resid   = (float*)alloc((size_t)MT * DM * 4);
  unsigned short* hs_bf = (unsigned short*)alloc((size_t)MT * DM * 2);
  float* xz      = (float*)alloc((size_t)MT * 2 * DI * 4);
  unsigned short* xcf0 = (unsigned short*)alloc((size_t)MT * DI * 2);
  unsigned short* xcf1 = (unsigned short*)alloc((size_t)MT * DI * 2);
  float* dbc0    = (float*)alloc((size_t)MT * XPN * 4);
  float* dbc1    = (float*)alloc((size_t)MT * XPN * 4);
  float* yraw0   = (float*)alloc((size_t)MT * DI * 4);
  float* yraw1   = (float*)alloc((size_t)MT * DI * 4);
  unsigned short* ybf = (unsigned short*)alloc((size_t)MT * DI * 2);
  float* ynorm   = (float*)alloc((size_t)B_ * DM * 4);

  const int GS = 4096;  // grid-stride block count for elementwise kernels

  // ---- one-time prep (runs every launch: deterministic) ----
  cvt_bf16_kernel<<<GS, 256, 0, stream>>>(in_proj,  wbf_inproj,  (size_t)DEPTH_ * 2 * DI * DM);
  cvt_bf16_kernel<<<GS, 256, 0, stream>>>(out_proj, wbf_outproj, (size_t)DEPTH_ * DM * DI);
  cvt_bf16_kernel<<<GS, 256, 0, stream>>>(patch_w,  wbf_patch,   (size_t)DM * 768);
  pack_xproj_kernel<<<GS, 256, 0, stream>>>(xproj_f, xproj_b, wbf_xproj);
  im2col_kernel<<<GS, 256, 0, stream>>>(x, im2c);

  // ---- patch embed GEMM: (6272 x 192) = im2col(6272x768) * patch_w(192x768)^T
  {
    dim3 g(DM / 64, (B_ * NP / 16 + 7) / 8);   // (3, 49)
    gemm_bf16_kernel<4><<<g, 256, 0, stream>>>(im2c, wbf_patch, featC, B_ * NP, DM, 768);
  }
  assemble_kernel<<<GS, 256, 0, stream>>>(featC, patch_b, cls_tok, pos_emb, hidden, resid);

  const unsigned gy = (unsigned)((MT / 16 + 7) / 8);  // 50 row-blocks
  for (int l = 0; l < DEPTH_; ++l) {
    prelayer_kernel<<<MT, 192, 0, stream>>>(resid, hidden, norm_w + (size_t)l * DM, hs_bf);

    // in_proj: (6304 x 768) = hs(6304x192) * W(768x192)^T
    { dim3 g((2 * DI) / 64, gy);
      gemm_bf16_kernel<4><<<g, 256, 0, stream>>>(hs_bf, wbf_inproj + (size_t)l * 2 * DI * DM,
                                                 xz, MT, 2 * DI, DM); }

    conv_kernel<<<GS, 256, 0, stream>>>(xz, conv_wf + (size_t)l * DI * DC,
                                        conv_bf_ + (size_t)l * DI, xcf0, 0);
    conv_kernel<<<GS, 256, 0, stream>>>(xz, conv_wb + (size_t)l * DI * DC,
                                        conv_bb_ + (size_t)l * DI, xcf1, 1);

    // x_proj: (6304 x 48) = xconv(6304x384) * Wpad(48x384)^T, both directions
    { dim3 g(1, gy);
      gemm_bf16_kernel<3><<<g, 256, 0, stream>>>(xcf0, wbf_xproj + (size_t)(l * 2 + 0) * XPN * DI,
                                                 dbc0, MT, XPN, DI);
      gemm_bf16_kernel<3><<<g, 256, 0, stream>>>(xcf1, wbf_xproj + (size_t)(l * 2 + 1) * XPN * DI,
                                                 dbc1, MT, XPN, DI); }

    scan_kernel<<<48, 256, 0, stream>>>(xcf0, dbc0,
        dtw_f + (size_t)l * DI * DTR, dtb_f + (size_t)l * DI,
        Alog_f + (size_t)l * DI * DS, D_f + (size_t)l * DI, yraw0);
    scan_kernel<<<48, 256, 0, stream>>>(xcf1, dbc1,
        dtw_b + (size_t)l * DI * DTR, dtb_b + (size_t)l * DI,
        Alog_b + (size_t)l * DI * DS, D_b + (size_t)l * DI, yraw1);

    combine_kernel<<<GS, 256, 0, stream>>>(yraw0, yraw1, xz, ybf);

    // out_proj: (6304 x 192) = y(6304x384) * W(192x384)^T -> hidden (next layer)
    { dim3 g(DM / 64, gy);
      gemm_bf16_kernel<4><<<g, 256, 0, stream>>>(ybf, wbf_outproj + (size_t)l * DM * DI,
                                                 hidden, MT, DM, DI); }
  }

  final_norm_kernel<<<B_, 192, 0, stream>>>(hidden, resid, normf_w, ynorm);
  head_kernel<<<(B_ * NCLS + 255) / 256, 256, 0, stream>>>(ynorm, head_w, head_b,
                                                           (float*)d_out);
}
